// ETNADemoPredictor_3650722201906
// MI455X (gfx1250) — matrix operand order repaired and checked by reading, not compiled, not run on hardware
//
#include <hip/hip_runtime.h>
#include <hip/hip_bf16.h>

typedef __attribute__((ext_vector_type(16))) __bf16 v16bf;
typedef __attribute__((ext_vector_type(8)))  float  v8f;

#define BATCH 4096
#define SEQL  200
#define DIM   64
#define LPAD  208          // 13 chunks of 16 rows
#define NCHUNK 13
#define NLAB  18

// ---------- bf16 helpers (round-to-nearest-even) ----------
__device__ __forceinline__ unsigned short f2bf(float f) {
    unsigned u = __float_as_uint(f);
    unsigned r = u + 0x7FFFu + ((u >> 16) & 1u);
    return (unsigned short)(r >> 16);
}
__device__ __forceinline__ float bf2f(unsigned short h) {
    return __uint_as_float(((unsigned)h) << 16);
}
__device__ __forceinline__ unsigned packbf(float f0, float f1) {
    return (unsigned)f2bf(f0) | ((unsigned)f2bf(f1) << 16);
}

struct Frag8 { unsigned w[8]; };
__device__ __forceinline__ v16bf load_frag(const unsigned* p) {
    Frag8 f;
#pragma unroll
    for (int i = 0; i < 8; ++i) f.w[i] = p[i];
    return __builtin_bit_cast(v16bf, f);
}

__device__ __forceinline__ v8f wmma_bf16(v16bf a, v16bf b, v8f c) {
    // D = A(16x32 bf16) * B(32x16 bf16) + C(16x16 f32)
    return __builtin_amdgcn_wmma_f32_16x16x32_bf16(
        false, a, false, b, (short)0, c, false, false);
}

// ============================================================
// Kernel 0: tran_W (3x64x64 fp32) -> bf16 WMMA B-fragment layout in ws.
// wsB[a][ks][nt][lane][v] : bf16 pair (K = kb, kb+1), column N = nt*16+(lane&15)
// B 32x16 layout: VGPR v -> K pair base = (v<4 ? 2v : 16+2(v-4)) + 8*half
//                 (same K order as the A fragment).
// ============================================================
__global__ void etna_prep_weights(const float* __restrict__ tranW,
                                  unsigned* __restrict__ wsB) {
    int tid = blockIdx.x * blockDim.x + threadIdx.x;
    if (tid >= 3 * 2 * 4 * 32 * 8) return;
    int v    = tid & 7;
    int lane = (tid >> 3) & 31;
    int nt   = (tid >> 8) & 3;
    int ks   = (tid >> 10) & 1;
    int a    = tid >> 11;
    int n    = nt * 16 + (lane & 15);
    int half = lane >> 4;
    int kb   = ks * 32 + ((v < 4) ? 2 * v : 16 + 2 * (v - 4)) + 8 * half;
    float f0 = tranW[a * 4096 + kb * 64 + n];
    float f1 = tranW[a * 4096 + (kb + 1) * 64 + n];
    wsB[tid] = packbf(f0, f1);
}

// ============================================================
// Main fused kernel: one workgroup (128 threads = 4 waves) per batch row.
// ============================================================
__global__ void __launch_bounds__(128)
etna_main(const int* __restrict__ x,
          const unsigned char* __restrict__ xmask,
          const float* __restrict__ ob,
          const float* __restrict__ emb,
          const float* __restrict__ tranb,
          const float* __restrict__ attW,
          const float* __restrict__ attb,
          const float* __restrict__ W0,
          const float* __restrict__ W1,
          const float* __restrict__ W2,
          const unsigned* __restrict__ wsB,
          float* __restrict__ out) {
    // A-fragment-layout embeddings: [chunk][lane][16] bf16 pairs (K 0..63)
    __shared__ unsigned       s_embA[NCHUNK * 32 * 16];   // 26624 B
    __shared__ unsigned short s_e[LPAD * 66];             // 27456 B (relu(e) bf16)
    __shared__ float          s_att[LPAD];                // 832 B
    __shared__ float          s_red[128];                 // 512 B
    __shared__ float          s_rep[3 * 64];              // 768 B
    __shared__ int            s_x[LPAD];                  // 832 B

    const int b    = blockIdx.x;
    const int tid  = threadIdx.x;
    const int lane = tid & 31;
    const int wave = tid >> 5;

    // ---- stage token ids ----
    for (int l = tid; l < LPAD; l += 128)
        s_x[l] = (l < SEQL) ? x[(size_t)b * SEQL + l] : 0;
    __syncthreads();

    // ---- gather embeddings directly into WMMA A-fragment layout ----
    // A 16x32 bf16 layout: lane->row m=lane&15, half=lane>>4,
    //   VGPR v: K pair base = ks*32 + (v<4 ? 2v : 16+2(v-4)) + 8*half
    for (int ent = tid; ent < NCHUNK * 32 * 16; ent += 128) {
        int chunk = ent >> 9;
        int r     = ent & 511;
        int ln    = r >> 4;
        int vv    = r & 15;
        int ks    = vv >> 3;
        int v     = vv & 7;
        int m     = ln & 15;
        int half  = ln >> 4;
        int l     = chunk * 16 + m;
        int kb    = ks * 32 + ((v < 4) ? 2 * v : 16 + 2 * (v - 4)) + 8 * half;
        float f0 = 0.f, f1 = 0.f;
        if (l < SEQL) {
            const float* p = emb + (size_t)s_x[l] * DIM + kb;
            f0 = p[0];
            f1 = p[1];
        }
        s_embA[ent] = packbf(f0, f1);
    }
    __syncthreads();

    // ---- per-attribute pipeline ----
    for (int a = 0; a < 3; ++a) {
        // === e = relu(embed @ tran_W[a] + b[a]) via WMMA, store bf16 in LDS ===
        for (int c = wave; c < NCHUNK; c += 4) {
            const unsigned* ab = &s_embA[(c * 32 + lane) * 16];
            v16bf A0 = load_frag(ab);        // K 0..31
            v16bf A1 = load_frag(ab + 8);    // K 32..63
            int n    = lane & 15;
            int half = lane >> 4;
#pragma unroll
            for (int nt = 0; nt < 4; ++nt) {
                const unsigned* b0 = wsB + (((a * 2 + 0) * 4 + nt) * 32 + lane) * 8;
                const unsigned* b1 = wsB + (((a * 2 + 1) * 4 + nt) * 32 + lane) * 8;
                v16bf B0 = load_frag(b0);
                v16bf B1 = load_frag(b1);
                v8f acc = {};
                acc = wmma_bf16(A0, B0, acc);
                acc = wmma_bf16(A1, B1, acc);
                int d = nt * 16 + n;
                float bias = tranb[a * 64 + d];
#pragma unroll
                for (int r = 0; r < 8; ++r) {
                    int l = c * 16 + r + 8 * half;   // C layout: VGPR r -> M=r+8*half
                    float val = fmaxf(acc[r] + bias, 0.f);
                    s_e[l * 66 + d] = f2bf(val);
                }
            }
        }
        __syncthreads();

        // === attention logits: att_u[l] = relu(e[l,:] . attW[a] + attb[a]) ===
        int l0 = tid, l1 = tid + 128;
        float u0 = 0.f, u1 = 0.f;
        int valid0 = 0, valid1 = 0;
        {
            valid0 = (l0 < SEQL) && (xmask[(size_t)b * SEQL + l0] != 0);
            float s = attb[a];
            for (int d = 0; d < 64; ++d)
                s += bf2f(s_e[l0 * 66 + d]) * attW[a * 64 + d];
            u0 = fmaxf(s, 0.f);
        }
        if (l1 < LPAD) {
            valid1 = (l1 < SEQL) && (xmask[(size_t)b * SEQL + l1] != 0);
            float s = attb[a];
            for (int d = 0; d < 64; ++d)
                s += bf2f(s_e[l1 * 66 + d]) * attW[a * 64 + d];
            u1 = fmaxf(s, 0.f);
        }

        // === masked softmax over valid prefix (workgroup reductions) ===
        float lm = -3.4e38f;
        if (valid0) lm = u0;
        if (valid1) lm = fmaxf(lm, u1);
        s_red[tid] = lm;
        __syncthreads();
#pragma unroll
        for (int s = 64; s > 0; s >>= 1) {
            if (tid < s) s_red[tid] = fmaxf(s_red[tid], s_red[tid + s]);
            __syncthreads();
        }
        float maxv = s_red[0];
        __syncthreads();

        float e0 = valid0 ? __expf(u0 - maxv) : 0.f;
        float e1 = valid1 ? __expf(u1 - maxv) : 0.f;
        s_red[tid] = e0 + e1;
        __syncthreads();
#pragma unroll
        for (int s = 64; s > 0; s >>= 1) {
            if (tid < s) s_red[tid] = s_red[tid] + s_red[tid + s];
            __syncthreads();
        }
        float inv = 1.f / s_red[0];
        __syncthreads();

        s_att[l0] = e0 * inv;
        if (l1 < LPAD) s_att[l1] = e1 * inv;
        __syncthreads();

        // === rep[a,d] = tanh( sum_l att[l] * e[l,d] ) ===
        {
            int d    = tid & 63;
            int part = tid >> 6;
            float acc = 0.f;
            for (int l = part * 104; l < part * 104 + 104; ++l)
                acc += s_att[l] * bf2f(s_e[l * 66 + d]);
            s_red[tid] = acc;
            __syncthreads();
            if (tid < 64)
                s_rep[a * 64 + tid] = tanhf(s_red[tid] + s_red[tid + 64]);
            __syncthreads();
        }
    }

    // ---- final interleaved projection: out[b,j] = (user_flat chunk . Wc) * ob ----
    // user_flat[b, k] = rep[k%3][k/3]   (d-major, attr-minor interleave)
    if (tid < NLAB) {
        int j = tid;
        int c, col, ncol;
        const float* W;
        if (j < 2)      { c = 0; col = j;     ncol = 2;  W = W0; }
        else if (j < 8) { c = 1; col = j - 2; ncol = 6;  W = W1; }
        else            { c = 2; col = j - 8; ncol = 10; W = W2; }
        float s = 0.f;
        for (int i = 0; i < 64; ++i) {
            int flat = c * 64 + i;
            s += s_rep[(flat % 3) * 64 + (flat / 3)] * W[i * ncol + col];
        }
        out[(size_t)b * NLAB + j] = s * ob[(size_t)b * NLAB + j];
    }
}

extern "C" void kernel_launch(void* const* d_in, const int* in_sizes, int n_in,
                              void* d_out, int out_size, void* d_ws, size_t ws_size,
                              hipStream_t stream) {
    (void)in_sizes; (void)n_in; (void)out_size; (void)ws_size;
    const int*           x     = (const int*)d_in[0];
    const unsigned char* xm    = (const unsigned char*)d_in[1];
    // d_in[2] = y (unused by the reference output)
    const float*         ob    = (const float*)d_in[3];
    const float*         emb   = (const float*)d_in[4];
    const float*         tranW = (const float*)d_in[5];
    const float*         tranb = (const float*)d_in[6];
    const float*         attW  = (const float*)d_in[7];
    const float*         attb  = (const float*)d_in[8];
    const float*         W0    = (const float*)d_in[9];
    const float*         W1    = (const float*)d_in[10];
    const float*         W2    = (const float*)d_in[11];
    float*               out   = (float*)d_out;
    unsigned*            wsB   = (unsigned*)d_ws;   // 24576 bytes used

    etna_prep_weights<<<24, 256, 0, stream>>>(tranW, wsB);
    etna_main<<<BATCH, 128, 0, stream>>>(x, xm, ob, emb, tranb, attW, attb,
                                         W0, W1, W2, wsB, out);
}
